// HexCompositionPredictor_16071767622245
// MI455X (gfx1250) — compile-verified, matrix operand order checked by reading, not run on hardware
//
#include <hip/hip_runtime.h>
#include <math.h>

// ---------------------------------------------------------------------------
// HexCompositionPredictor on MI455X (gfx1250).
// Dense GEMMs via v_wmma_f32_16x16x32_f16, one 16-node row tile per wave32.
// Weights pre-swizzled into B-fragment layout (f16) -> single b128-pair loads.
// LDS intermediates stored in A-fragment layout -> single ds b128-pair loads.
// Graph aggregation via global_atomic_add_f32 (fused matrix is L2-resident).
// ---------------------------------------------------------------------------

#define N_NODES 100000
#define N_EDGES 800000

typedef __attribute__((ext_vector_type(16))) _Float16 v16h;
typedef __attribute__((ext_vector_type(8)))  float    v8f;

__device__ __forceinline__ v8f wmma_f16(v16h a, v16h b, v8f c) {
  // (neg_a, A, neg_b, B, c_mod, C, reuse_a, reuse_b)
  return __builtin_amdgcn_wmma_f32_16x16x32_f16(false, a, false, b, (short)0, c,
                                                false, false);
}

__device__ __forceinline__ v8f zero8() { v8f z = {}; return z; }

// ---------------------------------------------------------------------------
// Fragment layout (per 16x16x32 WMMA step), wave32:
//   A (16x32, lane = row):    lanes 0-15 hold K {k0+0..7, k0+16..23},
//                             lanes16-31 hold K {k0+8..15, k0+24..31}
//   B (32x16, lane = column): same K split, column n = n0 + (lane&15)
//   C/D: lane = column n, VGPR i = row (i + 8*(lane>=16))
// Swizzled storage: fragment f stores 32 lanes x 16 halfs contiguously, so a
// fragment load is one 32-byte contiguous load per lane.
// ---------------------------------------------------------------------------

// ---- A fragment from global fp32 (row-guarded), float4-vectorized ----------
__device__ __forceinline__ v16h a_frag_global(const float* X, int ld, int row0,
                                              int k0) {
  const int lane = threadIdx.x & 31;
  const int row = row0 + (lane & 15);
  const int kb = k0 + ((lane & 16) ? 8 : 0);
  v16h a;
  if (row < N_NODES) {
    const float4* p = (const float4*)(X + row * ld + kb);
    const float4 v0 = p[0], v1 = p[1], v2 = p[4], v3 = p[5];
    a[0] = (_Float16)v0.x;  a[1] = (_Float16)v0.y;
    a[2] = (_Float16)v0.z;  a[3] = (_Float16)v0.w;
    a[4] = (_Float16)v1.x;  a[5] = (_Float16)v1.y;
    a[6] = (_Float16)v1.z;  a[7] = (_Float16)v1.w;
    a[8] = (_Float16)v2.x;  a[9] = (_Float16)v2.y;
    a[10] = (_Float16)v2.z; a[11] = (_Float16)v2.w;
    a[12] = (_Float16)v3.x; a[13] = (_Float16)v3.y;
    a[14] = (_Float16)v3.z; a[15] = (_Float16)v3.w;
  } else {
#pragma unroll
    for (int i = 0; i < 16; ++i) a[i] = (_Float16)0.f;
  }
  return a;
}

// Same, scaled by 1/max(deg,1) (GCN mean aggregation).
__device__ __forceinline__ v16h a_frag_scaled(const float* X, int ld, int row0,
                                              int k0, const float* deg) {
  const int lane = threadIdx.x & 31;
  const int row = row0 + (lane & 15);
  const int kb = k0 + ((lane & 16) ? 8 : 0);
  v16h a;
  if (row < N_NODES) {
    const float s = 1.f / fmaxf(deg[row], 1.f);
    const float4* p = (const float4*)(X + row * ld + kb);
    const float4 v0 = p[0], v1 = p[1], v2 = p[4], v3 = p[5];
    a[0] = (_Float16)(v0.x * s);  a[1] = (_Float16)(v0.y * s);
    a[2] = (_Float16)(v0.z * s);  a[3] = (_Float16)(v0.w * s);
    a[4] = (_Float16)(v1.x * s);  a[5] = (_Float16)(v1.y * s);
    a[6] = (_Float16)(v1.z * s);  a[7] = (_Float16)(v1.w * s);
    a[8] = (_Float16)(v2.x * s);  a[9] = (_Float16)(v2.y * s);
    a[10] = (_Float16)(v2.z * s); a[11] = (_Float16)(v2.w * s);
    a[12] = (_Float16)(v3.x * s); a[13] = (_Float16)(v3.y * s);
    a[14] = (_Float16)(v3.z * s); a[15] = (_Float16)(v3.w * s);
  } else {
#pragma unroll
    for (int i = 0; i < 16; ++i) a[i] = (_Float16)0.f;
  }
  return a;
}

// Masked target: tl*(1-m) + mask_token*m (ld == 64)
__device__ __forceinline__ v16h a_frag_masked(const float* tl, const int* mask,
                                              const float* mtok, int row0,
                                              int k0) {
  const int lane = threadIdx.x & 31;
  const int row = row0 + (lane & 15);
  const int kb = k0 + ((lane & 16) ? 8 : 0);
  v16h a;
  if (row < N_NODES) {
    const float4* p = (const float4*)(tl + row * 64 + kb);
    const int4* mp = (const int4*)(mask + row * 64 + kb);
    const float4* tp = (const float4*)(mtok + kb);
    const float4 v[4] = {p[0], p[1], p[4], p[5]};
    const int4 m[4] = {mp[0], mp[1], mp[4], mp[5]};
    const float4 t[4] = {tp[0], tp[1], tp[4], tp[5]};
#pragma unroll
    for (int j = 0; j < 4; ++j) {
      float mf;
      mf = (float)m[j].x; a[j * 4 + 0] = (_Float16)(v[j].x * (1.f - mf) + t[j].x * mf);
      mf = (float)m[j].y; a[j * 4 + 1] = (_Float16)(v[j].y * (1.f - mf) + t[j].y * mf);
      mf = (float)m[j].z; a[j * 4 + 2] = (_Float16)(v[j].z * (1.f - mf) + t[j].z * mf);
      mf = (float)m[j].w; a[j * 4 + 3] = (_Float16)(v[j].w * (1.f - mf) + t[j].w * mf);
    }
  } else {
#pragma unroll
    for (int i = 0; i < 16; ++i) a[i] = (_Float16)0.f;
  }
  return a;
}

// A fragment from per-wave LDS tile stored in fragment layout.
__device__ __forceinline__ v16h a_frag_lds_sw(const _Float16* T, int k0) {
  const int lane = threadIdx.x & 31;
  return *(const v16h*)(T + ((k0 >> 5) * 32 + lane) * 16);
}

// B fragment from pre-swizzled f16 weights. NT = Nout/16.
__device__ __forceinline__ v16h b_frag_pre(const _Float16* Wp, int NT, int k0,
                                           int nt) {
  const int lane = threadIdx.x & 31;
  return *(const v16h*)(Wp + (((k0 >> 5) * NT + nt) * 32 + lane) * 16);
}

// Store one column of a D tile (values val[i], rows mb+i, output col q) into a
// per-wave LDS tile laid out as A fragments for the next GEMM.
__device__ __forceinline__ _Float16* lds_store_base(_Float16* T, int q,
                                                    int mb) {
  const int kt = q >> 5, kk = q & 31;
  const int side = (kk >> 3) & 1;
  const int idx = (kk & 7) + ((kk >> 4) << 3);
  return T + (kt * 32 + side * 16 + mb) * 16 + idx;
}

// ---------------------------------------------------------------------------
// Weight pre-swizzle: fp32 W[K,N] row-major -> f16 B-fragment layout.
// out[((kt*NT + nt)*32 + lane)*16 + idx] = W[k, n]
// ---------------------------------------------------------------------------
__global__ __launch_bounds__(256) void prep_weights(const float* __restrict__ W,
                                                    _Float16* __restrict__ out,
                                                    int K, int N) {
  const int gid = blockIdx.x * blockDim.x + threadIdx.x;
  if (gid >= K * N) return;
  const int idx = gid & 15;
  const int lane = (gid >> 4) & 31;
  const int frag = gid >> 9;
  const int NT = N >> 4;
  const int kt = frag / NT, nt = frag - kt * NT;
  const int side = lane >> 4;
  const int n = nt * 16 + (lane & 15);
  const int kk = idx + side * 8 + (idx & 8);  // {0..7,16..23} / {8..15,24..31}
  const int k = kt * 32 + kk;
  out[gid] = (_Float16)W[k * N + n];
}

// ---------------------------------------------------------------------------
// Kernel 1: fused context + target encoder -> fused [N,192] fp32
// ---------------------------------------------------------------------------
__global__ __launch_bounds__(256) void encoder_kernel(
    const float* __restrict__ context, const float* __restrict__ tlog,
    const int* __restrict__ mask, const _Float16* __restrict__ w1p,
    const float* __restrict__ b1, const float* __restrict__ bng,
    const float* __restrict__ bnb, const float* __restrict__ bnm,
    const float* __restrict__ bnv, const _Float16* __restrict__ w2p,
    const float* __restrict__ b2, const _Float16* __restrict__ tewp,
    const float* __restrict__ teb, const float* __restrict__ mtok,
    float* __restrict__ fused) {
  __shared__ __align__(64) _Float16 hls[8][16 * 128];
  const int wave = threadIdx.x >> 5;
  const int row0 = (blockIdx.x * 8 + wave) * 16;
  if (row0 >= N_NODES) return;
  _Float16* hl = hls[wave];
  const int lane = threadIdx.x & 31;
  const int nl = lane & 15;
  const int mb = (lane & 16) ? 8 : 0;

  // ---- ctx layer 1: [16,128] @ [128,128] ----
  v8f acc[8];
#pragma unroll
  for (int t = 0; t < 8; ++t) acc[t] = zero8();
  for (int k0 = 0; k0 < 128; k0 += 32) {
    v16h a = a_frag_global(context, 128, row0, k0);
#pragma unroll
    for (int t = 0; t < 8; ++t)
      acc[t] = wmma_f16(a, b_frag_pre(w1p, 8, k0, t), acc[t]);
  }
  // eval-BN (bias folded) + ReLU -> LDS tile in A-fragment layout
#pragma unroll
  for (int t = 0; t < 8; ++t) {
    const int q = t * 16 + nl;
    const float sc = bng[q] * rsqrtf(bnv[q] + 1e-5f);
    const float sh = bnb[q] + (b1[q] - bnm[q]) * sc;
    _Float16* sb = lds_store_base(hl, q, mb);
#pragma unroll
    for (int i = 0; i < 8; ++i)
      sb[i * 16] = (_Float16)fmaxf(acc[t][i] * sc + sh, 0.f);
  }

  // ---- ctx layer 2: [16,128] @ [128,128] from LDS ----
#pragma unroll
  for (int t = 0; t < 8; ++t) acc[t] = zero8();
  for (int k0 = 0; k0 < 128; k0 += 32) {
    v16h a = a_frag_lds_sw(hl, k0);
#pragma unroll
    for (int t = 0; t < 8; ++t)
      acc[t] = wmma_f16(a, b_frag_pre(w2p, 8, k0, t), acc[t]);
  }
#pragma unroll
  for (int t = 0; t < 8; ++t) {
    const int n = t * 16 + nl;
    const float bb = b2[n];
#pragma unroll
    for (int i = 0; i < 8; ++i) {
      const int row = row0 + mb + i;
      if (row < N_NODES) fused[row * 192 + n] = fmaxf(acc[t][i] + bb, 0.f);
    }
  }

  // ---- target encoder: masked [16,64] @ [64,64] ----
  v8f at[4];
#pragma unroll
  for (int t = 0; t < 4; ++t) at[t] = zero8();
  for (int k0 = 0; k0 < 64; k0 += 32) {
    v16h a = a_frag_masked(tlog, mask, mtok, row0, k0);
#pragma unroll
    for (int t = 0; t < 4; ++t)
      at[t] = wmma_f16(a, b_frag_pre(tewp, 4, k0, t), at[t]);
  }
#pragma unroll
  for (int t = 0; t < 4; ++t) {
    const int n = t * 16 + nl;
    const float bb = teb[n];
#pragma unroll
    for (int i = 0; i < 8; ++i) {
      const int row = row0 + mb + i;
      if (row < N_NODES)
        fused[row * 192 + 128 + n] = fmaxf(at[t][i] + bb, 0.f);
    }
  }
}

// ---------------------------------------------------------------------------
// Edge kernels: degree + feature scatter-add
// ---------------------------------------------------------------------------
__global__ __launch_bounds__(256) void deg_kernel(const int* __restrict__ ei,
                                                  float* __restrict__ deg) {
  int e = blockIdx.x * blockDim.x + threadIdx.x;
  if (e < N_EDGES) atomicAdd(deg + ei[e], 1.f);
}

template <int F>
__global__ __launch_bounds__(256) void scatter_kernel(
    const int* __restrict__ ei, const float* __restrict__ x,
    float* __restrict__ agg) {
  const int CH = F / 4;
  long gid = (long)blockIdx.x * blockDim.x + threadIdx.x;
  if (gid >= (long)N_EDGES * CH) return;
  const int e = (int)(gid / CH);
  const int c = (int)(gid % CH);
  const int row = ei[e];
  const int col = ei[N_EDGES + e];
  const float4 v = ((const float4*)(x + (long)col * F))[c];
  float* dst = agg + (long)row * F + c * 4;
  atomicAdd(dst + 0, v.x);
  atomicAdd(dst + 1, v.y);
  atomicAdd(dst + 2, v.z);
  atomicAdd(dst + 3, v.w);
}

// ---------------------------------------------------------------------------
// GCN layer 1: h1 = relu((agg/deg) @ W[192,64] + b)
// ---------------------------------------------------------------------------
__global__ __launch_bounds__(256) void gcn1_kernel(
    const float* __restrict__ agg, const float* __restrict__ deg,
    const _Float16* __restrict__ Wp, const float* __restrict__ bias,
    float* __restrict__ h1) {
  const int wave = threadIdx.x >> 5;
  const int row0 = (blockIdx.x * 8 + wave) * 16;
  if (row0 >= N_NODES) return;
  const int lane = threadIdx.x & 31;
  const int nl = lane & 15;
  const int mb = (lane & 16) ? 8 : 0;
  v8f acc[4];
#pragma unroll
  for (int t = 0; t < 4; ++t) acc[t] = zero8();
  for (int k0 = 0; k0 < 192; k0 += 32) {
    v16h a = a_frag_scaled(agg, 192, row0, k0, deg);
#pragma unroll
    for (int t = 0; t < 4; ++t)
      acc[t] = wmma_f16(a, b_frag_pre(Wp, 4, k0, t), acc[t]);
  }
#pragma unroll
  for (int t = 0; t < 4; ++t) {
    const int n = t * 16 + nl;
    const float bb = bias[n];
#pragma unroll
    for (int i = 0; i < 8; ++i) {
      const int row = row0 + mb + i;
      if (row < N_NODES) h1[row * 64 + n] = fmaxf(acc[t][i] + bb, 0.f);
    }
  }
}

// ---------------------------------------------------------------------------
// GCN layer 2 + sigmoid(alpha) blend into hg
// ---------------------------------------------------------------------------
__global__ __launch_bounds__(256) void gcn2_kernel(
    const float* __restrict__ agg2, const float* __restrict__ deg,
    const _Float16* __restrict__ Wp, const float* __restrict__ bias,
    const float* __restrict__ alpha_p, int first, float* __restrict__ hg) {
  const int wave = threadIdx.x >> 5;
  const int row0 = (blockIdx.x * 8 + wave) * 16;
  if (row0 >= N_NODES) return;
  const int lane = threadIdx.x & 31;
  const int nl = lane & 15;
  const int mb = (lane & 16) ? 8 : 0;
  const float av = 1.f / (1.f + expf(-alpha_p[0]));
  const float coef = first ? av : (1.f - av);
  v8f acc[4];
#pragma unroll
  for (int t = 0; t < 4; ++t) acc[t] = zero8();
  for (int k0 = 0; k0 < 64; k0 += 32) {
    v16h a = a_frag_scaled(agg2, 64, row0, k0, deg);
#pragma unroll
    for (int t = 0; t < 4; ++t)
      acc[t] = wmma_f16(a, b_frag_pre(Wp, 4, k0, t), acc[t]);
  }
#pragma unroll
  for (int t = 0; t < 4; ++t) {
    const int n = t * 16 + nl;
    const float bb = bias[n];
#pragma unroll
    for (int i = 0; i < 8; ++i) {
      const int row = row0 + mb + i;
      if (row < N_NODES) {
        const float v = coef * (acc[t][i] + bb);
        if (first)
          hg[row * 64 + n] = v;
        else
          hg[row * 64 + n] += v;
      }
    }
  }
}

// ---------------------------------------------------------------------------
// Prediction head: [hg|fused] (K=256) -> 128 (BN,ReLU) -> 64 (ReLU) -> 64
// ---------------------------------------------------------------------------
__global__ __launch_bounds__(256) void head_kernel(
    const float* __restrict__ hg, const float* __restrict__ fused,
    const _Float16* __restrict__ w1p, const float* __restrict__ b1,
    const float* __restrict__ bng, const float* __restrict__ bnb,
    const float* __restrict__ bnm, const float* __restrict__ bnv,
    const _Float16* __restrict__ w2p, const float* __restrict__ b2,
    const _Float16* __restrict__ w3p, const float* __restrict__ b3,
    float* __restrict__ out) {
  __shared__ __align__(64) _Float16 h1s[8][16 * 128];
  __shared__ __align__(64) _Float16 h2s[8][16 * 64];
  const int wave = threadIdx.x >> 5;
  const int row0 = (blockIdx.x * 8 + wave) * 16;
  if (row0 >= N_NODES) return;
  _Float16* h1 = h1s[wave];
  _Float16* h2 = h2s[wave];
  const int lane = threadIdx.x & 31;
  const int nl = lane & 15;
  const int mb = (lane & 16) ? 8 : 0;

  // ph1: K=256 (cols 0..63 = hg, 64..255 = fused), Nout=128
  v8f acc[8];
#pragma unroll
  for (int t = 0; t < 8; ++t) acc[t] = zero8();
  for (int k0 = 0; k0 < 256; k0 += 32) {
    v16h a = (k0 < 64) ? a_frag_global(hg, 64, row0, k0)
                       : a_frag_global(fused, 192, row0, k0 - 64);
#pragma unroll
    for (int t = 0; t < 8; ++t)
      acc[t] = wmma_f16(a, b_frag_pre(w1p, 8, k0, t), acc[t]);
  }
#pragma unroll
  for (int t = 0; t < 8; ++t) {
    const int q = t * 16 + nl;
    const float sc = bng[q] * rsqrtf(bnv[q] + 1e-5f);
    const float sh = bnb[q] + (b1[q] - bnm[q]) * sc;
    _Float16* sb = lds_store_base(h1, q, mb);
#pragma unroll
    for (int i = 0; i < 8; ++i)
      sb[i * 16] = (_Float16)fmaxf(acc[t][i] * sc + sh, 0.f);
  }

  // ph2: K=128, Nout=64
  v8f a2[4];
#pragma unroll
  for (int t = 0; t < 4; ++t) a2[t] = zero8();
  for (int k0 = 0; k0 < 128; k0 += 32) {
    v16h a = a_frag_lds_sw(h1, k0);
#pragma unroll
    for (int t = 0; t < 4; ++t)
      a2[t] = wmma_f16(a, b_frag_pre(w2p, 4, k0, t), a2[t]);
  }
#pragma unroll
  for (int t = 0; t < 4; ++t) {
    const int q = t * 16 + nl;
    const float bb = b2[q];
    _Float16* sb = lds_store_base(h2, q, mb);
#pragma unroll
    for (int i = 0; i < 8; ++i)
      sb[i * 16] = (_Float16)fmaxf(a2[t][i] + bb, 0.f);
  }

  // ph3: K=64, Nout=64 -> output
  v8f a3[4];
#pragma unroll
  for (int t = 0; t < 4; ++t) a3[t] = zero8();
  for (int k0 = 0; k0 < 64; k0 += 32) {
    v16h a = a_frag_lds_sw(h2, k0);
#pragma unroll
    for (int t = 0; t < 4; ++t)
      a3[t] = wmma_f16(a, b_frag_pre(w3p, 4, k0, t), a3[t]);
  }
#pragma unroll
  for (int t = 0; t < 4; ++t) {
    const int n = t * 16 + nl;
    const float bb = b3[n];
#pragma unroll
    for (int i = 0; i < 8; ++i) {
      const int row = row0 + mb + i;
      if (row < N_NODES) out[row * 64 + n] = a3[t][i] + bb;
    }
  }
}

// ---------------------------------------------------------------------------
extern "C" void kernel_launch(void* const* d_in, const int* in_sizes, int n_in,
                              void* d_out, int out_size, void* d_ws,
                              size_t ws_size, hipStream_t stream) {
  (void)in_sizes; (void)n_in; (void)out_size; (void)ws_size;
  const float* context = (const float*)d_in[0];
  const float* tlog    = (const float*)d_in[1];
  const int*   mask    = (const int*)d_in[2];
  const int*   eis[2]  = {(const int*)d_in[3], (const int*)d_in[4]};
  const float* ce_w1 = (const float*)d_in[5];
  const float* ce_b1 = (const float*)d_in[6];
  const float* ce_bng = (const float*)d_in[7];
  const float* ce_bnb = (const float*)d_in[8];
  const float* ce_bnm = (const float*)d_in[9];
  const float* ce_bnv = (const float*)d_in[10];
  const float* ce_w2 = (const float*)d_in[11];
  const float* ce_b2 = (const float*)d_in[12];
  const float* te_w = (const float*)d_in[13];
  const float* te_b = (const float*)d_in[14];
  const float* mtok = (const float*)d_in[15];
  const float* g1w[2] = {(const float*)d_in[16], (const float*)d_in[20]};
  const float* g1b[2] = {(const float*)d_in[17], (const float*)d_in[21]};
  const float* g2w[2] = {(const float*)d_in[18], (const float*)d_in[22]};
  const float* g2b[2] = {(const float*)d_in[19], (const float*)d_in[23]};
  const float* alpha = (const float*)d_in[24];
  const float* ph_w1 = (const float*)d_in[25];
  const float* ph_b1 = (const float*)d_in[26];
  const float* ph_bng = (const float*)d_in[27];
  const float* ph_bnb = (const float*)d_in[28];
  const float* ph_bnm = (const float*)d_in[29];
  const float* ph_bnv = (const float*)d_in[30];
  const float* ph_w2 = (const float*)d_in[31];
  const float* ph_b2 = (const float*)d_in[32];
  const float* ph_w3 = (const float*)d_in[33];
  const float* ph_b3 = (const float*)d_in[34];

  // ---- workspace layout ----------------------------------------------------
  // [f16 swizzled weights: 114688 halfs = 224 KB] then fp32 buffers (~231 MB).
  _Float16* wf = (_Float16*)d_ws;
  _Float16* W_CE1 = wf;                    // 128*128
  _Float16* W_CE2 = W_CE1 + 16384;         // 128*128
  _Float16* W_TE  = W_CE2 + 16384;         // 64*64
  _Float16* W_G1[2]; _Float16* W_G2[2];
  W_G1[0] = W_TE + 4096;                   // 192*64
  W_G2[0] = W_G1[0] + 12288;               // 64*64
  W_G1[1] = W_G2[0] + 4096;                // 192*64
  W_G2[1] = W_G1[1] + 12288;               // 64*64
  _Float16* W_PH1 = W_G2[1] + 4096;        // 256*128
  _Float16* W_PH2 = W_PH1 + 32768;         // 128*64
  _Float16* W_PH3 = W_PH2 + 8192;          // 64*64

  float* fbase = (float*)d_ws + 57344;     // after 229376 bytes of f16
  float* fused = fbase;                              // N*192
  float* deg   = fused + (size_t)N_NODES * 192;      // N
  float* agg   = deg + N_NODES;                      // N*192
  float* h1    = agg + (size_t)N_NODES * 192;        // N*64
  float* agg2  = h1 + (size_t)N_NODES * 64;          // N*64
  float* hg    = agg2 + (size_t)N_NODES * 64;        // N*64

  // ---- pre-swizzle weights into B-fragment layout --------------------------
  auto prep = [&](const float* W, _Float16* out, int K, int N) {
    prep_weights<<<(K * N + 255) / 256, 256, 0, stream>>>(W, out, K, N);
  };
  prep(ce_w1, W_CE1, 128, 128);
  prep(ce_w2, W_CE2, 128, 128);
  prep(te_w, W_TE, 64, 64);
  prep(g1w[0], W_G1[0], 192, 64);
  prep(g2w[0], W_G2[0], 64, 64);
  prep(g1w[1], W_G1[1], 192, 64);
  prep(g2w[1], W_G2[1], 64, 64);
  prep(ph_w1, W_PH1, 256, 128);
  prep(ph_w2, W_PH2, 128, 64);
  prep(ph_w3, W_PH3, 64, 64);

  const int NBLK = (N_NODES + 127) / 128;  // 8 waves x 16 rows per block

  encoder_kernel<<<NBLK, 256, 0, stream>>>(
      context, tlog, mask, W_CE1, ce_b1, ce_bng, ce_bnb, ce_bnm, ce_bnv,
      W_CE2, ce_b2, W_TE, te_b, mtok, fused);

  for (int tw = 0; tw < 2; ++tw) {
    hipMemsetAsync(deg, 0, (size_t)N_NODES * sizeof(float), stream);
    hipMemsetAsync(agg, 0, (size_t)N_NODES * 192 * sizeof(float), stream);
    hipMemsetAsync(agg2, 0, (size_t)N_NODES * 64 * sizeof(float), stream);
    deg_kernel<<<(N_EDGES + 255) / 256, 256, 0, stream>>>(eis[tw], deg);
    scatter_kernel<192>
        <<<((long)N_EDGES * 48 + 255) / 256, 256, 0, stream>>>(eis[tw], fused,
                                                               agg);
    gcn1_kernel<<<NBLK, 256, 0, stream>>>(agg, deg, W_G1[tw], g1b[tw], h1);
    scatter_kernel<64>
        <<<((long)N_EDGES * 16 + 255) / 256, 256, 0, stream>>>(eis[tw], h1,
                                                               agg2);
    gcn2_kernel<<<NBLK, 256, 0, stream>>>(agg2, deg, W_G2[tw], g2b[tw], alpha,
                                          tw == 0 ? 1 : 0, hg);
  }

  head_kernel<<<NBLK, 256, 0, stream>>>(hg, fused, W_PH1, ph_b1, ph_bng,
                                        ph_bnb, ph_bnm, ph_bnv, W_PH2, ph_b2,
                                        W_PH3, ph_b3, (float*)d_out);
}
